// MoE_14353780703304
// MI455X (gfx1250) — matrix operand order repaired and checked by reading, not compiled, not run on hardware
//
#include <hip/hip_runtime.h>
#include <hip/hip_bf16.h>

#define E_ 8
#define C_ 768
#define F_ 3072
#define TILE_M 32
#define XPITCH 776   // 768 + 8 pad (16B-aligned rows, conflict-free strided reads)
#define HPITCH 136   // 128 + 8 pad
#define FCHUNK 128

// packed-tile geometry: tiles are 32(K) x 16(N) bf16, 512 elems, laid out
// [e][ktile][ntile][lane(32)][elem(16)] so each lane reads 32 contiguous bytes.
#define W1_KT (C_ / 32)   // 24
#define W1_NT (F_ / 16)   // 192
#define W2_KT (F_ / 32)   // 96
#define W2_NT (C_ / 16)   // 48

typedef __attribute__((ext_vector_type(16))) __bf16 v16bf;
typedef __attribute__((ext_vector_type(8)))  float  v8f;

union ATile {
    v16bf bf;
    unsigned short u[16];
    uint4 q[2];
};

__device__ __forceinline__ unsigned short f2bf(float f) {
    unsigned int u = __float_as_uint(f);
    u += 0x7FFFu + ((u >> 16) & 1u);   // round-to-nearest-even
    return (unsigned short)(u >> 16);
}

// ---------------------------------------------------------------- zero init
__global__ void moe_zero_kernel(float* __restrict__ out, int n, int* __restrict__ cnt) {
    int i = blockIdx.x * blockDim.x + threadIdx.x;
    if (i < E_) cnt[i] = 0;
    int stride = gridDim.x * blockDim.x;
    for (int p = i; p < n; p += stride) out[p] = 0.0f;
}

// ---------------------------------------------------------------- weight pre-pack
// src: fp32 [E][K][Ncols] row-major. dst: bf16 packed tiles per B-operand layout:
// lane L = 16*half + lr holds column (nt*16+lr), K = kt*32 + 16*half + e, e=0..15.
__global__ void moe_pack_kernel(const float* __restrict__ src,
                                unsigned short* __restrict__ dst,
                                int K, int Ncols, int tiles_total) {
    int idx = blockIdx.x * blockDim.x + threadIdx.x;   // one thread per (tile, lane)
    if (idx >= tiles_total * 32) return;
    int lane = idx & 31;
    int tile = idx >> 5;
    int ktc = K / 32, ntc = Ncols / 16;
    int e   = tile / (ktc * ntc);
    int rem = tile - e * (ktc * ntc);
    int kt  = rem / ntc;
    int nt  = rem - kt * ntc;
    int half = lane >> 4, lr = lane & 15;

    const float* s = src + (size_t)e * K * Ncols
                         + (size_t)(kt * 32 + 16 * half) * Ncols + (nt * 16 + lr);
    unsigned short tmp[16];
#pragma unroll
    for (int q = 0; q < 16; ++q) tmp[q] = f2bf(s[(size_t)q * Ncols]);
    uint4* d = (uint4*)(dst + (size_t)tile * 512 + lane * 16);
    d[0] = *(uint4*)&tmp[0];
    d[1] = *(uint4*)&tmp[8];
}

// ---------------------------------------------------------------- gating + routing
__global__ void moe_gate_kernel(const float* __restrict__ x,
                                const float* __restrict__ Wg,
                                const float* __restrict__ bg,
                                int N, int* __restrict__ cnt,
                                int* __restrict__ tok_list,
                                float* __restrict__ wgt_list) {
    __shared__ float wg_lds[C_ * E_];
    __shared__ float bg_lds[E_];
    for (int p = threadIdx.x; p < C_ * E_; p += blockDim.x) wg_lds[p] = Wg[p];
    if (threadIdx.x < E_) bg_lds[threadIdx.x] = bg[threadIdx.x];
    __syncthreads();

    int t = blockIdx.x * blockDim.x + threadIdx.x;
    if (t >= N) return;

    float acc[E_];
#pragma unroll
    for (int e = 0; e < E_; ++e) acc[e] = bg_lds[e];
    const float* xr = x + (size_t)t * C_;
    for (int c = 0; c < C_; ++c) {
        float xv = xr[c];
#pragma unroll
        for (int e = 0; e < E_; ++e) acc[e] = fmaf(xv, wg_lds[c * E_ + e], acc[e]);
    }
    // top-2 (lowest index wins ties, matching lax.top_k)
    int i0 = 0; float g0 = acc[0];
#pragma unroll
    for (int e = 1; e < E_; ++e) if (acc[e] > g0) { g0 = acc[e]; i0 = e; }
    int i1 = -1; float g1 = -3.4e38f;
#pragma unroll
    for (int e = 0; e < E_; ++e) if (e != i0 && acc[e] > g1) { g1 = acc[e]; i1 = e; }
    float ex  = __expf(g1 - g0);
    float inv = 1.0f / (1.0f + ex);
    float w0 = inv, w1 = ex * inv;

    int p0 = atomicAdd(&cnt[i0], 1);
    tok_list[i0 * N + p0] = t;  wgt_list[i0 * N + p0] = w0;
    int p1 = atomicAdd(&cnt[i1], 1);
    tok_list[i1 * N + p1] = t;  wgt_list[i1 * N + p1] = w1;
}

// ---------------------------------------------------------------- fused expert FFN
template <bool PACKED>
__global__ __launch_bounds__(512, 1)
void moe_expert_kernel(const float* __restrict__ x,
                       const float* __restrict__ W1,
                       const float* __restrict__ b1,
                       const float* __restrict__ W2,
                       const float* __restrict__ b2,
                       const unsigned short* __restrict__ W1p,
                       const unsigned short* __restrict__ W2p,
                       const int* __restrict__ cnt,
                       const int* __restrict__ tok_list,
                       const float* __restrict__ wgt_list,
                       float* __restrict__ out, int N) {
    const int e   = blockIdx.y;
    const int n_e = cnt[e];
    const int row0 = blockIdx.x * TILE_M;
    if (row0 >= n_e) return;   // uniform early-exit, EXEC stays all-ones for WMMA

    __shared__ __align__(16) unsigned short xlds[TILE_M * XPITCH]; // 48.5 KB bf16 X tile
    __shared__ __align__(16) unsigned short hlds[TILE_M * HPITCH]; // 8.5 KB bf16 H chunk
    __shared__ int   toklds[TILE_M];
    __shared__ float wlds[TILE_M];

    const int tid  = threadIdx.x;
    const int lane = tid & 31;
    const int wave = tid >> 5;          // 0..15
    const int half = lane >> 4;         // 0/1
    const int lr   = lane & 15;

    if (tid < TILE_M) {
        int idx = row0 + tid;
        if (idx < n_e) { toklds[tid] = tok_list[e * N + idx]; wlds[tid] = wgt_list[e * N + idx]; }
        else           { toklds[tid] = 0;                      wlds[tid] = 0.0f; }
    }
    __syncthreads();

    // Stage gathered X rows as bf16 into LDS (coalesced along C)
    for (int p = tid; p < TILE_M * C_; p += 512) {
        int r = p / C_;
        int c = p - r * C_;
        xlds[r * XPITCH + c] = f2bf(x[(size_t)toklds[r] * C_ + c]);
    }
    __syncthreads();

    const int hm = wave & 1, hn = wave >> 1;   // phase-1 tile of H (2 M-tiles x 8 N-tiles)
    const int mt = wave & 1, g  = wave >> 1;   // phase-2: 6 of 48 column tiles, M-tile mt

    v8f acc[6];
#pragma unroll
    for (int j = 0; j < 6; ++j) {
#pragma unroll
        for (int v = 0; v < 8; ++v) acc[j][v] = 0.0f;
    }

    const float*          W1e  = W1  + (size_t)e * C_ * F_;
    const float*          W2e  = W2  + (size_t)e * F_ * C_;
    const unsigned short* W1pe = W1p + (size_t)e * W1_KT * W1_NT * 512;
    const unsigned short* W2pe = W2p + (size_t)e * W2_KT * W2_NT * 512;

    for (int fc = 0; fc < F_ / FCHUNK; ++fc) {
        const int fbase = fc * FCHUNK;

        // -------- phase 1: Hc = relu(X @ W1[:, fbase:fbase+128] + b1)
        v8f hacc;
#pragma unroll
        for (int v = 0; v < 8; ++v) hacc[v] = 0.0f;

        const int fcol  = fbase + 16 * hn + lr;
        const int arow1 = 16 * hm + lr;

        if (PACKED && fc + 1 < F_ / FCHUNK) {
            // prefetch next F-chunk's first packed W1 tile for this wave
            __builtin_prefetch(W1pe + ((size_t)0 * W1_NT + ((fc + 1) * 8 + hn)) * 512 + lane * 16, 0, 3);
        }

#pragma unroll 4
        for (int kk = 0; kk < C_ / 32; ++kk) {
            ATile a, b;
            // A operand, contiguous k order (same as packed B): lane row = lane&15; K = 16*half+e (e=0..15)
            a.q[0] = *(const uint4*)&xlds[arow1 * XPITCH + 32 * kk + 16 * half];
            a.q[1] = *(const uint4*)&xlds[arow1 * XPITCH + 32 * kk + 16 * half + 8];
            if (PACKED) {
                const unsigned short* bp =
                    W1pe + ((size_t)kk * W1_NT + (fc * 8 + hn)) * 512 + lane * 16;
                b.q[0] = *(const uint4*)bp;
                b.q[1] = *(const uint4*)(bp + 8);
            } else {
                const float* w1p = W1e + (size_t)(32 * kk + 16 * half) * F_ + fcol;
#pragma unroll
                for (int q = 0; q < 16; ++q) b.u[q] = f2bf(w1p[(size_t)q * F_]);
            }
            hacc = __builtin_amdgcn_wmma_f32_16x16x32_bf16(
                false, a.bf, false, b.bf, (short)0, hacc, false, false);
        }
        float b1v = b1[(size_t)e * F_ + fcol];
#pragma unroll
        for (int v = 0; v < 8; ++v) {
            float h = fmaxf(hacc[v] + b1v, 0.0f);        // bias + ReLU
            int row = 16 * hm + 8 * half + v;            // C/D layout: M = v + 8*half
            hlds[row * HPITCH + 16 * hn + lr] = f2bf(h);
        }
        __syncthreads();

        // -------- phase 2: acc += Hc @ W2[fbase:fbase+128, :]
        const int arow2 = 16 * mt + lr;
#pragma unroll
        for (int ks = 0; ks < FCHUNK / 32; ++ks) {
            ATile a;
            a.q[0] = *(const uint4*)&hlds[arow2 * HPITCH + 32 * ks + 16 * half];
            a.q[1] = *(const uint4*)&hlds[arow2 * HPITCH + 32 * ks + 16 * half + 8];
            const int fk = fbase + 32 * ks + 16 * half;
#pragma unroll
            for (int j = 0; j < 6; ++j) {
                ATile b;
                if (PACKED) {
                    const unsigned short* bp =
                        W2pe + ((size_t)(fc * 4 + ks) * W2_NT + (g * 6 + j)) * 512 + lane * 16;
                    b.q[0] = *(const uint4*)bp;
                    b.q[1] = *(const uint4*)(bp + 8);
                } else {
                    const int col = 16 * (g * 6 + j) + lr;
                    const float* w2p = W2e + (size_t)fk * C_ + col;
#pragma unroll
                    for (int q = 0; q < 16; ++q) b.u[q] = f2bf(w2p[(size_t)q * C_]);
                }
                acc[j] = __builtin_amdgcn_wmma_f32_16x16x32_bf16(
                    false, a.bf, false, b.bf, (short)0, acc[j], false, false);
            }
        }
        __syncthreads();   // Hc reused next iteration
    }

    // -------- epilogue: out[token] += w * (acc + b2)
#pragma unroll
    for (int j = 0; j < 6; ++j) {
        const int col = 16 * (g * 6 + j) + lr;
        float b2v = b2[(size_t)e * C_ + col];
#pragma unroll
        for (int v = 0; v < 8; ++v) {
            int row = 16 * mt + 8 * half + v;
            float w = wlds[row];
            if (w != 0.0f) {
                atomicAdd(&out[(size_t)toklds[row] * C_ + col], w * (acc[j][v] + b2v));
            }
        }
    }
}

// ---------------------------------------------------------------- launch
extern "C" void kernel_launch(void* const* d_in, const int* in_sizes, int n_in,
                              void* d_out, int out_size, void* d_ws, size_t ws_size,
                              hipStream_t stream) {
    const float* x  = (const float*)d_in[0];
    const float* Wg = (const float*)d_in[1];
    const float* bg = (const float*)d_in[2];
    const float* W1 = (const float*)d_in[3];
    const float* b1 = (const float*)d_in[4];
    const float* W2 = (const float*)d_in[5];
    const float* b2 = (const float*)d_in[6];
    float* out = (float*)d_out;

    const int N = in_sizes[0] / C_;   // 8192 tokens

    // workspace: [0..63] expert counters | tok_list int[E*N] | wgt_list float[E*N]
    //            | (256B aligned) packed bf16 W1 | packed bf16 W2
    int*   cnt      = (int*)d_ws;
    int*   tok_list = (int*)((char*)d_ws + 64);
    float* wgt_list = (float*)((char*)d_ws + 64 + (size_t)E_ * N * sizeof(int));

    size_t off = 64 + (size_t)E_ * N * (sizeof(int) + sizeof(float));
    off = (off + 255) & ~(size_t)255;
    const size_t wbytes = (size_t)E_ * C_ * F_ * sizeof(unsigned short); // 37.75 MB each
    const bool packed = (ws_size >= off + 2 * wbytes);
    unsigned short* W1p = (unsigned short*)((char*)d_ws + off);
    unsigned short* W2p = (unsigned short*)((char*)d_ws + off + wbytes);

    moe_zero_kernel<<<1024, 256, 0, stream>>>(out, N * C_, cnt);
    moe_gate_kernel<<<(N + 255) / 256, 256, 0, stream>>>(x, Wg, bg, N, cnt, tok_list, wgt_list);

    dim3 grid((N + TILE_M - 1) / TILE_M, E_);
    if (packed) {
        const int tiles1 = E_ * W1_KT * W1_NT;   // 36864
        const int tiles2 = E_ * W2_KT * W2_NT;   // 36864
        moe_pack_kernel<<<(tiles1 * 32 + 255) / 256, 256, 0, stream>>>(W1, W1p, C_, F_, tiles1);
        moe_pack_kernel<<<(tiles2 * 32 + 255) / 256, 256, 0, stream>>>(W2, W2p, F_, C_, tiles2);
        moe_expert_kernel<true><<<grid, 512, 0, stream>>>(
            x, W1, b1, W2, b2, W1p, W2p, cnt, tok_list, wgt_list, out, N);
    } else {
        moe_expert_kernel<false><<<grid, 512, 0, stream>>>(
            x, W1, b1, W2, b2, W1p, W2p, cnt, tok_list, wgt_list, out, N);
    }
}
